// EncoderLayer_84241488544069
// MI455X (gfx1250) — matrix operand order repaired and checked by reading, not compiled, not run on hardware
//
#include <hip/hip_runtime.h>
#include <hip/hip_bf16.h>

#define DIM_   1024
#define FFN_   4096
#define HEADS_ 16
#define HD_    64
#define Bb_    2
#define Ss_    2048
#define MROWS  (Bb_*Ss_)

typedef __attribute__((ext_vector_type(16))) __bf16 v16bf;
typedef __attribute__((ext_vector_type(8)))  __bf16 v8bf;
typedef __attribute__((ext_vector_type(8)))  float  v8f;

static __device__ __forceinline__ v16bf cat8(v8bf lo, v8bf hi) {
  union { v16bf v; v8bf h[2]; } u; u.h[0] = lo; u.h[1] = hi; return u.v;
}
static __device__ __forceinline__ v8bf ld8(const __bf16* p) { return *(const v8bf*)p; }
static __device__ __forceinline__ v8f zero8() {
  v8f z;
#pragma unroll
  for (int i = 0; i < 8; ++i) z[i] = 0.f;
  return z;
}

// ---------------- elementwise converters ----------------
__global__ void cvt_f32_bf16(const float* __restrict__ in, __bf16* __restrict__ out, int n) {
  int i = blockIdx.x * 256 + threadIdx.x;
  if (i < n) out[i] = (__bf16)in[i];
}

// W[K,N] f32 -> Wt[N,K] bf16 (transpose so GEMM B-fragments are contiguous per lane)
__global__ void cvt_transpose_bf16(const float* __restrict__ W, __bf16* __restrict__ Wt,
                                   int K, int N) {
  int i = blockIdx.x * 256 + threadIdx.x;
  if (i >= K * N) return;
  int k = i / N, n = i % N;
  Wt[(size_t)n * K + k] = (__bf16)W[i];
}

// ---------------- WMMA GEMM: C[M,N] = (A[M,K](bf16,row) @ Bt[N,K](bf16) + bias) * outScale --------
// one wave computes a 64x64 tile; 16 v_wmma_f32_16x16x32_bf16 per K=32 step.
// 32-bit element offsets from uniform bases -> saddr-form global loads.
// VT: outB stored as V^T layout [B,HEADS,HD,S] (attention B-operand).
template <bool OUTF, bool OUTB, bool RELU, bool VT>
__global__ __launch_bounds__(256) void gemm_wmma_bf16(
    const __bf16* __restrict__ A, const __bf16* __restrict__ Bt,
    const float* __restrict__ bias, float* __restrict__ outF,
    __bf16* __restrict__ outB, int M, int N, int K, float outScale) {
  const int lane = threadIdx.x & 31;
  const int wave = threadIdx.x >> 5;
  const int l15  = lane & 15;
  const int hiL  = (lane >> 4) & 1;
  const int tilesN = N >> 6;
  const int tile = blockIdx.x * 8 + wave;
  if (tile >= (M >> 6) * tilesN) return;
  const int m0 = (tile / tilesN) * 64;
  const int n0 = (tile % tilesN) * 64;

  unsigned aoff[4], boff[4];
#pragma unroll
  for (int f = 0; f < 4; ++f) {
    aoff[f] = (unsigned)(m0 + f * 16 + l15) * K + hiL * 16;  // A: K {0..15}/{16..31}
    boff[f] = (unsigned)(n0 + f * 16 + l15) * K + hiL * 16;  // B: K {0..15}/{16..31}
  }

  v8f acc[4][4];
#pragma unroll
  for (int a = 0; a < 4; ++a)
#pragma unroll
    for (int b = 0; b < 4; ++b) acc[a][b] = zero8();

  for (int k = 0; k < K; k += 32) {
    v16bf af[4], bf_[4];
#pragma unroll
    for (int f = 0; f < 4; ++f) {
      af[f]  = cat8(ld8(A + (aoff[f] + k)), ld8(A + (aoff[f] + k + 8)));
      bf_[f] = cat8(ld8(Bt + (boff[f] + k)), ld8(Bt + (boff[f] + k + 8)));
    }
    __builtin_prefetch(A + (aoff[0] + k + 32), 0, 3);
    __builtin_prefetch(Bt + (boff[0] + k + 32), 0, 3);
#pragma unroll
    for (int a = 0; a < 4; ++a)
#pragma unroll
      for (int b = 0; b < 4; ++b)
        acc[a][b] = __builtin_amdgcn_wmma_f32_16x16x32_bf16(
            false, af[a], false, bf_[b], (short)0, acc[a][b], false, false);
  }

#pragma unroll
  for (int b = 0; b < 4; ++b) {
    const int n = n0 + b * 16 + l15;
    const float bv = bias[n];
#pragma unroll
    for (int a = 0; a < 4; ++a) {
#pragma unroll
      for (int r = 0; r < 8; ++r) {
        const int m = m0 + a * 16 + r + hiL * 8;
        float v = (acc[a][b][r] + bv) * outScale;
        if constexpr (RELU) v = fmaxf(v, 0.f);
        if constexpr (OUTF) outF[(unsigned)m * N + n] = v;
        if constexpr (OUTB) {
          if constexpr (VT) {
            const int hh = n >> 6, dd = n & (HD_ - 1);
            const int bb = m >> 11, ss = m & (Ss_ - 1);
            outB[(((unsigned)bb * HEADS_ + hh) * HD_ + dd) * Ss_ + ss] = (__bf16)v;
          } else {
            outB[(unsigned)m * N + n] = (__bf16)v;
          }
        }
      }
    }
  }
}

// ---------------- flash attention: barrier-free, one wave = 16 query rows ----------------
// Q pre-scaled by 1/sqrt(DIM); K row-major [B,S,D]; V transposed [B,H,HD,S].
// K fragment (lane=key, d contiguous) and V fragment (lane=d, keys contiguous) load
// straight from global; the 8 waves of a block share one (b,h) so loads hit L0/L2.
__global__ __launch_bounds__(256) void attn_wmma(
    const __bf16* __restrict__ Qb, const __bf16* __restrict__ Kb,
    const __bf16* __restrict__ Vt, const int* __restrict__ mask,
    __bf16* __restrict__ Ob) {
  const int tid = threadIdx.x;
  const int lane = tid & 31, wave = tid >> 5;
  const int l15 = lane & 15, hiL = (lane >> 4) & 1;
  const int gw = blockIdx.x * 8 + wave;      // global wave id
  const int qt = gw & (Ss_ / 16 - 1);        // 128 q-tiles of 16 rows per (b,h)
  const int hb = gw >> 7;
  const int h = hb & (HEADS_ - 1), b = hb >> 4;

  __shared__ __bf16 Ps[8][16 * 32];          // per-wave P transpose scratch

  const unsigned qoff = (unsigned)(b * Ss_ + qt * 16 + l15) * DIM_ + h * HD_ + hiL * 16;
  const v16bf q0 = cat8(ld8(Qb + qoff),      ld8(Qb + qoff + 8));
  const v16bf q1 = cat8(ld8(Qb + qoff + 32), ld8(Qb + qoff + 40));

  v8f O[4];
  float mrow[8], lrow[8];
#pragma unroll
  for (int f = 0; f < 4; ++f) O[f] = zero8();
#pragma unroll
  for (int r = 0; r < 8; ++r) { mrow[r] = -1e30f; lrow[r] = 0.f; }

  const unsigned khead = (unsigned)b * Ss_ * DIM_ + h * HD_ + hiL * 16;   // + key*DIM_
  const unsigned vhead = (unsigned)(b * HEADS_ + h) * HD_ * Ss_ + hiL * 16;

  for (int kt = 0; kt < Ss_; kt += 32) {
    // scores: two 16x16 C-fragments (keys kt..kt+15, kt+16..kt+31)
    v8f s0 = zero8(), s1 = zero8();
    {
      const unsigned kr0 = khead + (unsigned)(kt + l15) * DIM_;
      const unsigned kr1 = khead + (unsigned)(kt + 16 + l15) * DIM_;
      v16bf k00 = cat8(ld8(Kb + kr0),      ld8(Kb + kr0 + 8));       // d 0..31
      v16bf k01 = cat8(ld8(Kb + kr0 + 32), ld8(Kb + kr0 + 40));      // d 32..63
      v16bf k10 = cat8(ld8(Kb + kr1),      ld8(Kb + kr1 + 8));
      v16bf k11 = cat8(ld8(Kb + kr1 + 32), ld8(Kb + kr1 + 40));
      s0 = __builtin_amdgcn_wmma_f32_16x16x32_bf16(false, q0, false, k00, (short)0, s0, false, false);
      s0 = __builtin_amdgcn_wmma_f32_16x16x32_bf16(false, q1, false, k01, (short)0, s0, false, false);
      s1 = __builtin_amdgcn_wmma_f32_16x16x32_bf16(false, q0, false, k10, (short)0, s1, false, false);
      s1 = __builtin_amdgcn_wmma_f32_16x16x32_bf16(false, q1, false, k11, (short)0, s1, false, false);
    }
    const int mk0 = mask[b * Ss_ + kt + l15];
    const int mk1 = mask[b * Ss_ + kt + 16 + l15];
#pragma unroll
    for (int r = 0; r < 8; ++r) {
      s0[r] = mk0 ? s0[r] : -1e30f;
      s1[r] = mk1 ? s1[r] : -1e30f;
    }
    // online softmax: row stats live across the 16 lanes holding each row
    float corr[8];
#pragma unroll
    for (int r = 0; r < 8; ++r) {
      float t = fmaxf(s0[r], s1[r]);
#pragma unroll
      for (int off = 1; off < 16; off <<= 1) t = fmaxf(t, __shfl_xor(t, off, 32));
      const float nm = fmaxf(mrow[r], t);
      corr[r] = __expf(mrow[r] - nm);
      mrow[r] = nm;
      s0[r] = __expf(s0[r] - nm);
      s1[r] = __expf(s1[r] - nm);
      float ps = s0[r] + s1[r];
#pragma unroll
      for (int off = 1; off < 16; off <<= 1) ps += __shfl_xor(ps, off, 32);
      lrow[r] = lrow[r] * corr[r] + ps;
    }
#pragma unroll
    for (int f = 0; f < 4; ++f)
#pragma unroll
      for (int r = 0; r < 8; ++r) O[f][r] *= corr[r];

    // P: C-layout -> bf16 A-layout via per-wave LDS scratch (DS ops in-order per wave)
    __bf16* pw = Ps[wave];
#pragma unroll
    for (int r = 0; r < 8; ++r) {
      const int mr = r + hiL * 8;
      pw[mr * 32 + l15]      = (__bf16)s0[r];
      pw[mr * 32 + 16 + l15] = (__bf16)s1[r];
    }
    const v16bf pa = cat8(ld8(pw + l15 * 32 + hiL * 16), ld8(pw + l15 * 32 + hiL * 16 + 8));
    // V fragments (lane=d, 16 contiguous keys) direct from transposed V
#pragma unroll
    for (int f = 0; f < 4; ++f) {
      const unsigned vo = vhead + (unsigned)(f * 16 + l15) * Ss_ + kt;
      const v16bf vf = cat8(ld8(Vt + vo), ld8(Vt + vo + 8));
      O[f] = __builtin_amdgcn_wmma_f32_16x16x32_bf16(false, pa, false, vf, (short)0, O[f], false, false);
    }
  }

#pragma unroll
  for (int f = 0; f < 4; ++f)
#pragma unroll
    for (int r = 0; r < 8; ++r) {
      const int mr = r + hiL * 8;
      const float ov = O[f][r] / lrow[r];
      Ob[(unsigned)(b * Ss_ + qt * 16 + mr) * DIM_ + h * HD_ + f * 16 + l15] = (__bf16)ov;
    }
}

// ---------------- fused residual + layernorm: out = LN(x + y + y2) ----------------
__global__ __launch_bounds__(256) void add_ln(
    const float* __restrict__ x, const float* __restrict__ y, const float* __restrict__ y2,
    const float* __restrict__ gamma, const float* __restrict__ beta,
    float* __restrict__ out) {
  const int row = blockIdx.x, t = threadIdx.x;
  const size_t base = (size_t)row * DIM_;
  float v[4], s = 0.f, ss = 0.f;
#pragma unroll
  for (int i = 0; i < 4; ++i) {
    const int c = t + i * 256;
    const float a = x[base + c] + y[base + c] + y2[base + c];
    v[i] = a; s += a; ss += a * a;
  }
  __shared__ float rs[256], rss[256];
  rs[t] = s; rss[t] = ss;
  __syncthreads();
  for (int o = 128; o > 0; o >>= 1) {
    if (t < o) { rs[t] += rs[t + o]; rss[t] += rss[t + o]; }
    __syncthreads();
  }
  const float mu = rs[0] * (1.f / DIM_);
  const float var = rss[0] * (1.f / DIM_) - mu * mu;
  const float inv = rsqrtf(var + 1e-5f);
#pragma unroll
  for (int i = 0; i < 4; ++i) {
    const int c = t + i * 256;
    out[base + c] = (v[i] - mu) * inv * gamma[c] + beta[c];
  }
}

extern "C" void kernel_launch(void* const* d_in, const int* in_sizes, int n_in,
                              void* d_out, int out_size, void* d_ws, size_t ws_size,
                              hipStream_t stream) {
  (void)in_sizes; (void)n_in; (void)out_size; (void)ws_size;
  const float* x    = (const float*)d_in[0];
  const int*   msk  = (const int*)d_in[1];
  const float* Wq   = (const float*)d_in[2];  const float* bq = (const float*)d_in[3];
  const float* Wk   = (const float*)d_in[4];  const float* bk = (const float*)d_in[5];
  const float* Wv   = (const float*)d_in[6];  const float* bv = (const float*)d_in[7];
  const float* Wo   = (const float*)d_in[8];  const float* bo = (const float*)d_in[9];
  const float* W1   = (const float*)d_in[10]; const float* b1 = (const float*)d_in[11];
  const float* W2   = (const float*)d_in[12]; const float* b2 = (const float*)d_in[13];
  const float* gamma = (const float*)d_in[14];
  const float* beta  = (const float*)d_in[15];
  float* out = (float*)d_out;

  char* ws = (char*)d_ws;
  size_t off = 0;
  auto take = [&](size_t bytes) { char* p = ws + off; off += (bytes + 255) & ~size_t(255); return p; };
  const size_t MD = (size_t)MROWS * DIM_;
  __bf16* xb    = (__bf16*)take(MD * 2);
  __bf16* WqT   = (__bf16*)take((size_t)DIM_ * DIM_ * 2);
  __bf16* WkT   = (__bf16*)take((size_t)DIM_ * DIM_ * 2);
  __bf16* WvT   = (__bf16*)take((size_t)DIM_ * DIM_ * 2);
  __bf16* WoT   = (__bf16*)take((size_t)DIM_ * DIM_ * 2);
  __bf16* W1T   = (__bf16*)take((size_t)DIM_ * FFN_ * 2);   // [FFN, DIM]
  __bf16* W2T   = (__bf16*)take((size_t)FFN_ * DIM_ * 2);   // [DIM, FFN]
  __bf16* Qb    = (__bf16*)take(MD * 2);
  __bf16* Kb    = (__bf16*)take(MD * 2);
  __bf16* Vtb   = (__bf16*)take(MD * 2);                    // V^T [B,H,HD,S]
  __bf16* attnb = (__bf16*)take(MD * 2);
  float*  y     = (float*)take(MD * 4);
  __bf16* yb    = (__bf16*)take(MD * 2);
  float*  y2    = (float*)take(MD * 4);
  __bf16* hbuf  = Qb;  // reuse Qb/Kb/Vtb/attnb region (32MB) for FFN hidden after attention

  const float qscale = 0.03125f;  // 1/sqrt(DIM) = 1/32 (reference scales by full model dim)

  // 1) conversions / weight transposes
  cvt_f32_bf16<<<(int)((MD + 255) / 256), 256, 0, stream>>>(x, xb, (int)MD);
  const int wde = DIM_ * DIM_;
  cvt_transpose_bf16<<<(wde + 255) / 256, 256, 0, stream>>>(Wq, WqT, DIM_, DIM_);
  cvt_transpose_bf16<<<(wde + 255) / 256, 256, 0, stream>>>(Wk, WkT, DIM_, DIM_);
  cvt_transpose_bf16<<<(wde + 255) / 256, 256, 0, stream>>>(Wv, WvT, DIM_, DIM_);
  cvt_transpose_bf16<<<(wde + 255) / 256, 256, 0, stream>>>(Wo, WoT, DIM_, DIM_);
  const int wfe = DIM_ * FFN_;
  cvt_transpose_bf16<<<(wfe + 255) / 256, 256, 0, stream>>>(W1, W1T, DIM_, FFN_);
  cvt_transpose_bf16<<<(wfe + 255) / 256, 256, 0, stream>>>(W2, W2T, FFN_, DIM_);

  // 2) Q/K/V projections (Q pre-scaled; V stored transposed for the attention B-operand)
  const int gQKV = (MROWS / 64) * (DIM_ / 64) / 8;  // 128
  gemm_wmma_bf16<false, true, false, false><<<gQKV, 256, 0, stream>>>(
      xb, WqT, bq, nullptr, Qb, MROWS, DIM_, DIM_, qscale);
  gemm_wmma_bf16<false, true, false, false><<<gQKV, 256, 0, stream>>>(
      xb, WkT, bk, nullptr, Kb, MROWS, DIM_, DIM_, 1.f);
  gemm_wmma_bf16<false, true, false, true><<<gQKV, 256, 0, stream>>>(
      xb, WvT, bv, nullptr, Vtb, MROWS, DIM_, DIM_, 1.f);

  // 3) attention (barrier-free; 8 waves/block share one (b,h))
  const int gAtt = Bb_ * HEADS_ * (Ss_ / 16) / 8;   // 512
  attn_wmma<<<gAtt, 256, 0, stream>>>(Qb, Kb, Vtb, msk, attnb);

  // 4) output projection -> y (f32 for residual) + yb (bf16 for FFN input)
  gemm_wmma_bf16<true, true, false, false><<<gQKV, 256, 0, stream>>>(
      attnb, WoT, bo, y, yb, MROWS, DIM_, DIM_, 1.f);

  // 5) FFN1 (ReLU) -> hidden bf16
  const int gF1 = (MROWS / 64) * (FFN_ / 64) / 8;   // 512
  gemm_wmma_bf16<false, true, true, false><<<gF1, 256, 0, stream>>>(
      yb, W1T, b1, nullptr, hbuf, MROWS, FFN_, DIM_, 1.f);

  // 6) FFN2 -> y2 f32
  gemm_wmma_bf16<true, false, false, false><<<gQKV, 256, 0, stream>>>(
      hbuf, W2T, b2, y2, nullptr, MROWS, DIM_, FFN_, 1.f);

  // 7) out = LN(x + y + y2)
  add_ln<<<MROWS, 256, 0, stream>>>(x, y, y2, gamma, beta, out);
}
